// TauJastrow_48627619726072
// MI455X (gfx1250) — matrix operand order repaired and checked by reading, not compiled, not run on hardware
//
#include <hip/hip_runtime.h>

// ---------------------------------------------------------------------------
// TauJastrow on MI455X (gfx1250, wave32). All three MLPs (phi/psi/rho) run on
// v_wmma_f32_16x16x32_f16 (f16 in, f32 accumulate). One 128-thread workgroup
// (4 wave32) owns 16 batch elements entirely in LDS.
//
// Round-2 changes:
//  * All weights stored in LDS in B-fragment order with zero padding baked in:
//    every WMMA B operand = 2x contiguous ds_load_b128, no guards/branches.
//  * A-tile loads vectorized (aligned 8 x f16 chunks).
//  * Branchless GELU (tanh-form via v_exp_f32 + v_rcp_f32) replaces the
//    branchy OCML erff; native __logf/__expf/__sinf/__cosf in scalar paths.
//  * Pair gate stashed in padding column 6 of the input tile (B rows >=6 are
//    zero, so A padding columns only need to be finite).
// ---------------------------------------------------------------------------

typedef __attribute__((ext_vector_type(16))) _Float16 v16h;
typedef __attribute__((ext_vector_type(8)))  _Float16 v8h;
typedef __attribute__((ext_vector_type(8)))  float    v8f;

#define WMMA_F16(a, b, c) \
  __builtin_amdgcn_wmma_f32_16x16x32_f16(false, (a), false, (b), (short)0, (c), false, false)

namespace {

constexpr int kBTot    = 16384;  // batch
constexpr int kN       = 16;     // particles
constexpr int kD       = 3;
constexpr int kH       = 64;
constexpr int kDL      = 5;
constexpr int kTE      = 16;
constexpr int kNF      = 8;
constexpr int kRhoIn   = 2 * kDL + 2 + kTE;  // 28
constexpr int kPairs   = kN * (kN - 1) / 2;  // 120
constexpr int kBpWG    = 16;                 // batches per workgroup
constexpr int kThreads = 128;                // 4 waves
constexpr int kPsiTilesPerWave = (kBpWG * kPairs / 16) / 4;  // 30

constexpr float kSqrtOmega = 1.0f;           // OMEGA = 1.0
constexpr float kEps2      = 0.04f;          // (0.2 * a_ho)^2
constexpr float kRg2       = 0.09f;          // (0.3 * a_ho)^2
constexpr float kFeps      = 1.1920929e-7f;  // float32 eps

struct Smem {
  // Weights in B-fragment order: [(kt*NT + nt)*32 + lane]*16 + e, zero-padded.
  __attribute__((aligned(16))) _Float16 w_phi0p[4 * 32 * 16];  // 1 kt x 4 nt
  __attribute__((aligned(16))) _Float16 w_phi1p[8 * 32 * 16];  // 2 kt x 4 nt
  __attribute__((aligned(16))) _Float16 w_phi2p[2 * 32 * 16];  // 2 kt x 1 nt
  __attribute__((aligned(16))) _Float16 w_psi0p[4 * 32 * 16];
  __attribute__((aligned(16))) _Float16 w_psi1p[8 * 32 * 16];
  __attribute__((aligned(16))) _Float16 w_psi2p[2 * 32 * 16];
  __attribute__((aligned(16))) _Float16 w_rho0p[4 * 32 * 16];
  __attribute__((aligned(16))) _Float16 w_rho1p[8 * 32 * 16];
  __attribute__((aligned(16))) _Float16 w_rho2p[2 * 32 * 16];
  // Per-wave staging. inbuf[0] doubles as the rho input tile in phase 4.
  __attribute__((aligned(16))) _Float16 inbuf[4][16 * 32];  // 16xK in (K pad 32)
  __attribute__((aligned(16))) _Float16 hbuf[4][16 * 64];   // 16x64 activations
  // Biases (f32); layer-2 biases zero-padded to 16.
  float b_phi0[kH], b_phi1[kH], b_phi2[16];
  float b_psi0[kH], b_psi1[kH], b_psi2[16];
  float b_rho0[kH], b_rho1[kH], b_rho2[16];
  // Per-WG data.
  float xs[kBpWG * kN * kD];
  float phi_sum[kBpWG][8];
  float psi_sum[kBpWG][8];
  float s1_sum[kBpWG];
  float cusp[kBpWG];
  unsigned char pii[kPairs], pjj[kPairs];
};

// Branchless GELU (tanh form): x * sigmoid(2*sqrt(2/pi)*(x + 0.044715 x^3)).
__device__ __forceinline__ float gelu_f(float v) {
  const float u = 1.5957691216057308f * v * (1.0f + 0.044715f * v * v);
  return v * __builtin_amdgcn_rcpf(1.0f + __expf(-u));
}

// A fragment (16x32 f16) from row-major [16][strideK] LDS tile, K-tile kt.
// Lane l: M = l&15; lanes 0-15 hold K={0..7,16..23}, lanes 16-31 K={8..15,24..31}.
__device__ __forceinline__ v16h load_A_tile(const _Float16* t, int lane, int kt,
                                            int strideK) {
  const int m  = lane & 15;
  const int hi = lane >> 4;
  const _Float16* p = t + m * strideK + kt * 32 + hi * 8;
  const v8h a0 = *(const v8h*)(p);
  const v8h a1 = *(const v8h*)(p + 16);
  v16h r;
#pragma unroll
  for (int j = 0; j < 8; ++j) { r[j] = a0[j]; r[8 + j] = a1[j]; }
  return r;
}

// B fragment from pre-permuted weights: 32 contiguous bytes per lane.
__device__ __forceinline__ v16h load_frag(const _Float16* base, int lane, int tile) {
  const v8h* p = (const v8h*)(base + (tile * 32 + lane) * 16);
  const v8h lo = p[0];
  const v8h hi = p[1];
  v16h r;
#pragma unroll
  for (int j = 0; j < 8; ++j) { r[j] = lo[j]; r[8 + j] = hi[j]; }
  return r;
}

// Store a C tile (16x16 f32) with bias + GELU into a 16x64 f16 LDS tile.
// C layout: VGPR i, lane l -> M = i + 8*(l>>4), N = (l&15) + nt*16.
__device__ __forceinline__ void store_h_gelu(_Float16* t, int lane, int nt, v8f c,
                                             const float* bias) {
  const int n  = (lane & 15) + nt * 16;
  const int mb = (lane >> 4) * 8;
  const float bn = bias[n];
#pragma unroll
  for (int i = 0; i < 8; ++i) {
    t[(mb + i) * 64 + n] = (_Float16)gelu_f(c[i] + bn);
  }
}

// Loop-invariant fragments of one network (input layer + output layer).
struct NetFrags {
  v16h b0[4];  // layer0, kt=0, nt=0..3
  v16h b2[2];  // layer2, kt=0..1, nt=0
};

__device__ __forceinline__ void preload_net(NetFrags& f, const _Float16* w0p,
                                            const _Float16* w2p, int lane) {
#pragma unroll
  for (int nt = 0; nt < 4; ++nt) f.b0[nt] = load_frag(w0p, lane, nt);
#pragma unroll
  for (int kt = 0; kt < 2; ++kt) f.b2[kt] = load_frag(w2p, lane, kt);
}

// One 16-row tile through KIN->64(gelu)->64(gelu)->NOUT. Returns final C tile
// (layer-2 bias not yet added). 14 WMMAs.
__device__ __forceinline__ v8f mlp_tile(const _Float16* inb, _Float16* hb, int lane,
                                        const NetFrags& f, const _Float16* w1p,
                                        const float* b0, const float* b1) {
  {  // layer 0
    v16h a = load_A_tile(inb, lane, 0, 32);
#pragma unroll
    for (int nt = 0; nt < 4; ++nt) {
      v8f c = {};
      c = WMMA_F16(a, f.b0[nt], c);
      store_h_gelu(hb, lane, nt, c, b0);
    }
  }
  {  // layer 1: preload A to regs, then overwrite hb (same-wave DS is in-order)
    v16h a0 = load_A_tile(hb, lane, 0, 64);
    v16h a1 = load_A_tile(hb, lane, 1, 64);
#pragma unroll
    for (int nt = 0; nt < 4; ++nt) {
      v8f c = {};
      c = WMMA_F16(a0, load_frag(w1p, lane, nt), c);
      c = WMMA_F16(a1, load_frag(w1p, lane, 4 + nt), c);
      store_h_gelu(hb, lane, nt, c, b1);
    }
  }
  // layer 2
  v16h a0 = load_A_tile(hb, lane, 0, 64);
  v16h a1 = load_A_tile(hb, lane, 1, 64);
  v8f c = {};
  c = WMMA_F16(a0, f.b2[0], c);
  c = WMMA_F16(a1, f.b2[1], c);
  return c;
}

// Stage weights (row-major f32 [KV x NV]) into fragment order with zero pad.
template <int KT, int NT, int KV, int NV>
__device__ __forceinline__ void stage_wfrag(_Float16* dst, const float* src, int tid) {
  constexpr int total = KT * NT * 32 * 16;
  for (int i = tid; i < total; i += kThreads) {
    const int e  = i & 15;
    const int ln = (i >> 4) & 31;
    const int ti = i >> 9;  // fragment tile index = kt*NT + nt
    const int nt = ti % NT;
    const int kt = ti / NT;
    const int n  = (ln & 15) + nt * 16;
    const int k  = kt * 32 + (ln >> 4) * 8 + (e < 8 ? e : 8 + e);
    float v = 0.0f;
    if (k < KV && n < NV) v = src[k * NV + n];
    dst[i] = (_Float16)v;
  }
}

}  // namespace

__global__ __launch_bounds__(kThreads) void taujastrow_kernel(
    const float* __restrict__ x, const float* __restrict__ tau,
    const float* __restrict__ freqs,
    const float* __restrict__ phi_w0, const float* __restrict__ phi_b0,
    const float* __restrict__ phi_w1, const float* __restrict__ phi_b1,
    const float* __restrict__ phi_w2, const float* __restrict__ phi_b2,
    const float* __restrict__ psi_w0, const float* __restrict__ psi_b0,
    const float* __restrict__ psi_w1, const float* __restrict__ psi_b1,
    const float* __restrict__ psi_w2, const float* __restrict__ psi_b2,
    const float* __restrict__ rho_w0, const float* __restrict__ rho_b0,
    const float* __restrict__ rho_w1, const float* __restrict__ rho_b1,
    const float* __restrict__ rho_w2, const float* __restrict__ rho_b2,
    const float* __restrict__ tp_w0, const float* __restrict__ tp_b0,
    const float* __restrict__ tp_w1, const float* __restrict__ tp_b1,
    float* __restrict__ out) {
  __shared__ Smem s;

  const int tid  = threadIdx.x;
  const int lane = tid & 31;
  const int wave = tid >> 5;
  const int b0g  = blockIdx.x * kBpWG;

  // -------------------- phase 0: stage weights + inputs ---------------------
  for (int i = tid; i < kBpWG * kN * kD; i += kThreads)
    s.xs[i] = x[(size_t)b0g * kN * kD + i] * kSqrtOmega;

  stage_wfrag<1, 4, kD, kH>(s.w_phi0p, phi_w0, tid);
  stage_wfrag<2, 4, kH, kH>(s.w_phi1p, phi_w1, tid);
  stage_wfrag<2, 1, kH, kDL>(s.w_phi2p, phi_w2, tid);
  stage_wfrag<1, 4, 6, kH>(s.w_psi0p, psi_w0, tid);
  stage_wfrag<2, 4, kH, kH>(s.w_psi1p, psi_w1, tid);
  stage_wfrag<2, 1, kH, kDL>(s.w_psi2p, psi_w2, tid);
  stage_wfrag<1, 4, kRhoIn, kH>(s.w_rho0p, rho_w0, tid);
  stage_wfrag<2, 4, kH, kH>(s.w_rho1p, rho_w1, tid);
  stage_wfrag<2, 1, kH, 1>(s.w_rho2p, rho_w2, tid);

  for (int i = tid; i < kH; i += kThreads) {
    s.b_phi0[i] = phi_b0[i]; s.b_phi1[i] = phi_b1[i];
    s.b_psi0[i] = psi_b0[i]; s.b_psi1[i] = psi_b1[i];
    s.b_rho0[i] = rho_b0[i]; s.b_rho1[i] = rho_b1[i];
  }
  if (tid < 16) {
    s.b_phi2[tid] = (tid < kDL) ? phi_b2[tid] : 0.0f;
    s.b_psi2[tid] = (tid < kDL) ? psi_b2[tid] : 0.0f;
    s.b_rho2[tid] = (tid < 1)   ? rho_b2[tid] : 0.0f;
    s.s1_sum[tid] = 0.0f;
    s.cusp[tid]   = 0.0f;
#pragma unroll
    for (int n = 0; n < 8; ++n) { s.phi_sum[tid][n] = 0.0f; s.psi_sum[tid][n] = 0.0f; }
  }
  if (tid < kPairs) {  // triu(k=1) pair table
    int p = tid, i = 0, len = kN - 1;
    while (p >= len) { p -= len; ++i; --len; }
    s.pii[tid] = (unsigned char)i;
    s.pjj[tid] = (unsigned char)(i + 1 + p);
  }
  __syncthreads();

  _Float16* inb = s.inbuf[wave];
  _Float16* hb  = s.hbuf[wave];

  // One-time zero of the input tile: guarantees all padding columns are finite
  // (weight fragments are zero there, so any finite value contributes 0).
  for (int i = lane; i < 16 * 32; i += 32) inb[i] = (_Float16)0.0f;

  // -------------------- phase 1: psi over all pairs (WMMA) ------------------
  {
    NetFrags fr;
    preload_net(fr, s.w_psi0p, s.w_psi2p, lane);
    for (int tt = 0; tt < kPsiTilesPerWave; ++tt) {
      const int t = wave * kPsiTilesPerWave + tt;  // tile of 16 pairs
      if (lane < 16) {  // lanes 0-15 build one pair row each
        const int p  = t * 16 + lane;
        const int bl = p / kPairs;
        const int pr = p - bl * kPairs;
        const int pi = s.pii[pr], pj = s.pjj[pr];
        const float* xb = &s.xs[bl * kN * kD];
        const float dx = xb[pi * 3 + 0] - xb[pj * 3 + 0];
        const float dy = xb[pi * 3 + 1] - xb[pj * 3 + 1];
        const float dz = xb[pi * 3 + 2] - xb[pj * 3 + 2];
        const float r2  = dx * dx + dy * dy + dz * dz;
        const float r   = __builtin_amdgcn_sqrtf(r2 + kFeps);
        const float rt2 = r2 + kEps2;
        const float u   = rt2 * (1.0f / kEps2);
        const float s1  = __logf(1.0f + u);       // u >= 1: log1p == log(1+u)
        const float s2  = r2 * __builtin_amdgcn_rcpf(rt2);
        const float s3  = u * __expf(-u);
        _Float16* row = &inb[lane * 32];
        row[0] = (_Float16)s1;
        row[1] = (_Float16)s2;
        row[2] = (_Float16)s3;
        row[3] = (_Float16)__expf(-0.25f * s1);
        row[4] = (_Float16)__expf(-1.00f * s1);
        row[5] = (_Float16)__expf(-4.00f * s1);
        row[6] = (_Float16)(r2 * __builtin_amdgcn_rcpf(r2 + kRg2));  // gate
        atomicAdd(&s.s1_sum[bl], s1);
        const bool same = (pi < kN / 2) == (pj < kN / 2);
        const float gamma = same ? (1.0f / (kD + 1)) : (1.0f / (kD - 1));
        atomicAdd(&s.cusp[bl], gamma * r * __expf(-r));
      }
      v8f c = mlp_tile(inb, hb, lane, fr, s.w_psi1p, s.b_psi0, s.b_psi1);
      const int n = lane & 15;
      if (n < kDL) {
        const int mb = (lane >> 4) * 8;
#pragma unroll
        for (int i = 0; i < 8; ++i) {
          const int m = mb + i;
          const int p = t * 16 + m;
          const float g = (float)inb[m * 32 + 6];
          atomicAdd(&s.psi_sum[p / kPairs][n], (c[i] + s.b_psi2[n]) * g);
        }
      }
    }
  }

  // -------------------- phase 2: phi over all particles (WMMA) --------------
  {
    NetFrags fr;
    preload_net(fr, s.w_phi0p, s.w_phi2p, lane);
    for (int kk = 0; kk < kBpWG / 4; ++kk) {
      const int bl = wave * (kBpWG / 4) + kk;  // one batch = one 16-particle tile
      if (lane < 16) {
        const float* xp = &s.xs[(bl * kN + lane) * kD];
        _Float16* row = &inb[lane * 32];
        row[0] = (_Float16)xp[0];
        row[1] = (_Float16)xp[1];
        row[2] = (_Float16)xp[2];
      }
      v8f c = mlp_tile(inb, hb, lane, fr, s.w_phi1p, s.b_phi0, s.b_phi1);
      const int n = lane & 15;
      if (n < kDL) {
        const int mb = (lane >> 4) * 8;
#pragma unroll
        for (int i = 0; i < 8; ++i) atomicAdd(&s.phi_sum[bl][n], c[i] + s.b_phi2[n]);
      }
    }
  }
  __syncthreads();

  // -------------------- phase 3: assemble rho input (scalar) ----------------
  // rho input tile lives in inbuf[0] (free after phases 1-2).
  if (tid < kBpWG) {
    const int b = tid;
    float acc = 0.0f;
    const float* xb = &s.xs[b * kN * kD];
#pragma unroll
    for (int q = 0; q < kN * kD; ++q) acc += xb[q] * xb[q];
    const float r2m = acc * (1.0f / (kN * kD));

    const float tv = tau[b0g + b];
    float tf[2 * kNF + 1];
    tf[0] = tv;
#pragma unroll
    for (int k = 0; k < kNF; ++k) {
      const float ft = freqs[k] * tv;
      tf[1 + k]       = __sinf(ft);
      tf[1 + kNF + k] = __cosf(ft);
    }
    float hid[kTE];
#pragma unroll
    for (int h = 0; h < kTE; ++h) {
      float a = tp_b0[h];
      for (int k = 0; k < 2 * kNF + 1; ++k) a += tf[k] * tp_w0[k * kTE + h];
      hid[h] = gelu_f(a);
    }
    _Float16* row = &s.inbuf[0][b * 32];
#pragma unroll
    for (int n = 0; n < kDL; ++n)
      row[n] = (_Float16)(s.phi_sum[b][n] * (1.0f / kN));
#pragma unroll
    for (int n = 0; n < kDL; ++n)
      row[kDL + n] = (_Float16)(s.psi_sum[b][n] * (1.0f / kPairs));
    row[2 * kDL + 0] = (_Float16)r2m;
    row[2 * kDL + 1] = (_Float16)(s.s1_sum[b] * (1.0f / kPairs));
#pragma unroll
    for (int e = 0; e < kTE; ++e) {
      float a = tp_b1[e];
      for (int h = 0; h < kTE; ++h) a += hid[h] * tp_w1[h * kTE + e];
      row[2 * kDL + 2 + e] = (_Float16)a;
    }
    row[28] = row[29] = row[30] = row[31] = (_Float16)0.0f;
  }
  __syncthreads();

  // -------------------- phase 4: rho MLP + output (wave 0, WMMA) ------------
  if (wave == 0) {
    NetFrags fr;
    preload_net(fr, s.w_rho0p, s.w_rho2p, lane);
    v8f c = mlp_tile(s.inbuf[0], s.hbuf[0], lane, fr, s.w_rho1p, s.b_rho0, s.b_rho1);
    if ((lane & 15) == 0) {  // column 0 is the scalar rho output per batch row
      const int mb = (lane >> 4) * 8;
#pragma unroll
      for (int i = 0; i < 8; ++i) {
        const int b = mb + i;
        out[b0g + b] = c[i] + s.b_rho2[0] + s.cusp[b];
      }
    }
  }
}

extern "C" void kernel_launch(void* const* d_in, const int* in_sizes, int n_in,
                              void* d_out, int out_size, void* d_ws, size_t ws_size,
                              hipStream_t stream) {
  (void)in_sizes; (void)n_in; (void)out_size; (void)d_ws; (void)ws_size;
  const float* x      = (const float*)d_in[0];
  const float* tau    = (const float*)d_in[1];
  const float* freqs  = (const float*)d_in[2];
  const float* phi_w0 = (const float*)d_in[3];
  const float* phi_b0 = (const float*)d_in[4];
  const float* phi_w1 = (const float*)d_in[5];
  const float* phi_b1 = (const float*)d_in[6];
  const float* phi_w2 = (const float*)d_in[7];
  const float* phi_b2 = (const float*)d_in[8];
  const float* psi_w0 = (const float*)d_in[9];
  const float* psi_b0 = (const float*)d_in[10];
  const float* psi_w1 = (const float*)d_in[11];
  const float* psi_b1 = (const float*)d_in[12];
  const float* psi_w2 = (const float*)d_in[13];
  const float* psi_b2 = (const float*)d_in[14];
  const float* rho_w0 = (const float*)d_in[15];
  const float* rho_b0 = (const float*)d_in[16];
  const float* rho_w1 = (const float*)d_in[17];
  const float* rho_b1 = (const float*)d_in[18];
  const float* rho_w2 = (const float*)d_in[19];
  const float* rho_b2 = (const float*)d_in[20];
  const float* tp_w0  = (const float*)d_in[21];
  const float* tp_b0  = (const float*)d_in[22];
  const float* tp_w1  = (const float*)d_in[23];
  const float* tp_b1  = (const float*)d_in[24];

  const int grid = kBTot / kBpWG;  // 1024 workgroups x 128 threads
  taujastrow_kernel<<<grid, kThreads, 0, stream>>>(
      x, tau, freqs,
      phi_w0, phi_b0, phi_w1, phi_b1, phi_w2, phi_b2,
      psi_w0, psi_b0, psi_w1, psi_b1, psi_w2, psi_b2,
      rho_w0, rho_b0, rho_w1, rho_b1, rho_w2, rho_b2,
      tp_w0, tp_b0, tp_w1, tp_b1,
      (float*)d_out);
}
